// GAT_Block_28578712388225
// MI455X (gfx1250) — compile-verified, matrix operand order checked by reading, not run on hardware
//
#include <hip/hip_runtime.h>
#include <hip/hip_bf16.h>
#include <stdint.h>

typedef __attribute__((ext_vector_type(16))) __bf16 v16bf;
typedef __attribute__((ext_vector_type(8)))  float  v8f;

#define IN_C   128
#define HC     256      // HEADS * OUT_C
#define OUT_C  64
#define HEADS  4
#define NEG_SLOPE 0.2f

__device__ __forceinline__ unsigned short f2bf(float f) {
    unsigned u = __float_as_uint(f);
    u += 0x7FFFu + ((u >> 16) & 1u);   // round-to-nearest-even
    return (unsigned short)(u >> 16);
}
__device__ __forceinline__ float bf2f(unsigned short h) {
    return __uint_as_float(((unsigned)h) << 16);
}
// monotonic float<->uint transform for atomicMax on floats
__device__ __forceinline__ unsigned f2ord(float f) {
    unsigned u = __float_as_uint(f);
    return (u & 0x80000000u) ? ~u : (u | 0x80000000u);
}
__device__ __forceinline__ float ord2f(unsigned e) {
    return __uint_as_float((e & 0x80000000u) ? (e & 0x7FFFFFFFu) : ~e);
}

// ---------------- utility kernels ----------------
__global__ void zero_u32(unsigned* __restrict__ p, int n) {
    int i = blockIdx.x * blockDim.x + threadIdx.x;
    if (i < n) p[i] = 0u;
}

__global__ void f32_to_bf16_kernel(const float* __restrict__ in,
                                   unsigned short* __restrict__ out, int n) {
    int i = blockIdx.x * blockDim.x + threadIdx.x;
    if (i < n) out[i] = f2bf(in[i]);
}

// Pack B [K,N] f32 row-major into WMMA-B fragment order (bf16):
// tile (ks,nt) = 32 lanes * 16 elems; lane elem i: K = ks*32 + (i<8?0:16) + (lane>>4)*8 + (i&7),
// N = nt*16 + (lane&15).  Each lane's 16 elems are contiguous => 2x16B loads in GEMM.
__global__ void pack_b(const float* __restrict__ B, unsigned short* __restrict__ Bp,
                       int K, int N) {
    int o = blockIdx.x * blockDim.x + threadIdx.x;
    if (o >= K * N) return;
    int NT = N >> 4;
    int tile = o >> 9;        // /512
    int within = o & 511;
    int lane = within >> 4;
    int e16  = within & 15;
    int j = e16 >> 3;         // 0 or 1 (K half 0..7 vs 16..23)
    int e = e16 & 7;
    int ks = tile / NT;
    int nt = tile - ks * NT;
    int Kidx = ks * 32 + j * 16 + (lane >> 4) * 8 + e;
    int Ncol = nt * 16 + (lane & 15);
    Bp[o] = f2bf(B[(size_t)Kidx * N + Ncol]);
}

// ---------------- WMMA GEMM: C[M,N] = A[M,K] @ B[K,N] (+bias) ----------------
template<int K, int N>
__global__ __launch_bounds__(256) void gemm_wmma_bf16(
    const unsigned short* __restrict__ A,   // [M,K] bf16 row-major
    const unsigned short* __restrict__ Bp,  // packed B (see pack_b)
    const float* __restrict__ bias,         // [N] or nullptr
    unsigned short* __restrict__ outB,      // [M,N] bf16 or nullptr
    float* __restrict__ outF,               // [M,N] f32  or nullptr
    int M)
{
    constexpr int NT  = N / 16;
    constexpr int KS  = K / 32;
    constexpr int NTH = NT / 2;             // n-tiles per wave
    __shared__ __align__(16) unsigned short sA[64 * K];

    const int tid = threadIdx.x;
    const int blk = blockIdx.x;

    // cooperative 128-bit staged load of A tile (64 rows x K), OOB rows -> 0
    const int chunks = (64 * K) / 8;
    for (int i = tid; i < chunks; i += 256) {
        int row = (i * 8) / K;
        int col = (i * 8) % K;
        int g = blk * 64 + row;
        uint4 v = make_uint4(0u, 0u, 0u, 0u);
        if (g < M) v = *reinterpret_cast<const uint4*>(A + (size_t)g * K + col);
        *reinterpret_cast<uint4*>(&sA[row * K + col]) = v;
    }
    __syncthreads();

    const int lane  = tid & 31;
    const int wave  = tid >> 5;
    const int rtile = wave & 3;      // 4 row-tiles of 16
    const int ngrp  = wave >> 2;     // 2 wave-groups split the N tiles
    const int half  = lane >> 4;
    const int l15   = lane & 15;

    v8f acc[NTH];
    #pragma unroll
    for (int i = 0; i < NTH; ++i)
        acc[i] = v8f{0.f,0.f,0.f,0.f,0.f,0.f,0.f,0.f};

    union Frag { uint4 u[2]; v16bf v; };

    for (int ks = 0; ks < KS; ++ks) {
        Frag fa;
        const int arow = rtile * 16 + l15;
        const int aoff = arow * K + ks * 32 + half * 8;
        fa.u[0] = *reinterpret_cast<const uint4*>(&sA[aoff]);
        fa.u[1] = *reinterpret_cast<const uint4*>(&sA[aoff + 16]);
        #pragma unroll
        for (int nt = 0; nt < NTH; ++nt) {
            const int gnt = ngrp * NTH + nt;
            Frag fb;
            const size_t boff = ((size_t)(ks * NT + gnt) * 32 + lane) * 16;
            fb.u[0] = *reinterpret_cast<const uint4*>(Bp + boff);
            fb.u[1] = *reinterpret_cast<const uint4*>(Bp + boff + 8);
            acc[nt] = __builtin_amdgcn_wmma_f32_16x16x32_bf16(
                false, fa.v, false, fb.v, (short)0, acc[nt], false, false);
        }
    }

    // epilogue: C lane mapping -> row = base + (lane>>4)*8 + r, col = gnt*16 + (lane&15)
    #pragma unroll
    for (int nt = 0; nt < NTH; ++nt) {
        const int gnt = ngrp * NTH + nt;
        const int col = gnt * 16 + l15;
        const float bc = bias ? bias[col] : 0.0f;
        const int rbase = blk * 64 + rtile * 16 + half * 8;
        #pragma unroll
        for (int r = 0; r < 8; ++r) {
            const int grow = rbase + r;
            if (grow < M) {
                float v = acc[nt][r] + bc;
                if (outB) outB[(size_t)grow * N + col] = f2bf(v);
                if (outF) outF[(size_t)grow * N + col] = v;
            }
        }
    }
}

// ---------------- attention kernels ----------------
__global__ void attn_scores(const unsigned short* __restrict__ hb,   // [N,256] bf16
                            const float* __restrict__ a_src,         // [4,64]
                            const float* __restrict__ a_dst,         // [4,64]
                            float* __restrict__ als, float* __restrict__ ald, int Nn)
{
    int gid = blockIdx.x * blockDim.x + threadIdx.x;
    if (gid >= Nn * HEADS) return;
    int n = gid >> 2, h = gid & 3;
    const unsigned short* hp = hb + (size_t)n * HC + h * OUT_C;
    const float* as = a_src + h * OUT_C;
    const float* ad = a_dst + h * OUT_C;
    float s0 = 0.f, s1 = 0.f;
    #pragma unroll 8
    for (int c = 0; c < OUT_C; ++c) {
        float v = bf2f(hp[c]);
        s0 += v * as[c];
        s1 += v * ad[c];
    }
    als[gid] = s0;
    ald[gid] = s1;
}

__global__ void edge_max(const long long* __restrict__ ei, int E, int Nn,
                         const float* __restrict__ als, const float* __restrict__ ald,
                         unsigned* __restrict__ menc)
{
    int i = blockIdx.x * blockDim.x + threadIdx.x;
    if (i >= E + Nn) return;
    int src, dst;
    if (i < E) { src = (int)ei[i]; dst = (int)ei[E + i]; }
    else       { src = dst = i - E; }               // self-loops
    #pragma unroll
    for (int h = 0; h < HEADS; ++h) {
        float e = als[src * 4 + h] + ald[dst * 4 + h];
        e = e > 0.f ? e : NEG_SLOPE * e;
        atomicMax(&menc[dst * 4 + h], f2ord(e));
    }
}

// one wave per edge; lane l handles head l>>3, channels ((l&7)*8 .. +7)
__global__ __launch_bounds__(256) void edge_acc(
    const long long* __restrict__ ei, int E, int Nn,
    const float* __restrict__ als, const float* __restrict__ ald,
    const unsigned* __restrict__ menc, const unsigned short* __restrict__ hb,
    float* __restrict__ ssum, float* __restrict__ acc)
{
    int lane = threadIdx.x & 31;
    int edge = blockIdx.x * 8 + (threadIdx.x >> 5);
    if (edge >= E + Nn) return;
    int src, dst;
    if (edge < E) { src = (int)ei[edge]; dst = (int)ei[E + edge]; }
    else          { src = dst = edge - E; }
    int h  = lane >> 3;
    int c0 = (lane & 7) * 8;
    float e = als[src * 4 + h] + ald[dst * 4 + h];
    e = e > 0.f ? e : NEG_SLOPE * e;
    float m  = ord2f(menc[dst * 4 + h]);
    float ex = __expf(e - m);
    if ((lane & 7) == 0) atomicAdd(&ssum[dst * 4 + h], ex);
    const unsigned short* hp = hb + (size_t)src * HC + h * OUT_C + c0;
    float* ap = acc + (size_t)dst * HC + h * OUT_C + c0;
    #pragma unroll
    for (int c = 0; c < 8; ++c)
        atomicAdd(&ap[c], ex * bf2f(hp[c]));
}

__global__ void finalize1(const float* __restrict__ acc, const float* __restrict__ ssum,
                          const float* __restrict__ b1, unsigned short* __restrict__ f1b, int Nn)
{
    int gid = blockIdx.x * blockDim.x + threadIdx.x;
    if (gid >= Nn * HC) return;
    int n = gid >> 8, j = gid & 255, h = j >> 6;
    float v = acc[gid] / (ssum[n * 4 + h] + 1e-16f) + b1[j];
    f1b[gid] = f2bf(v);
}

__global__ void finalize2(const float* __restrict__ acc, const float* __restrict__ ssum,
                          const float* __restrict__ b2, const float* __restrict__ xfc,
                          float* __restrict__ out, int Nn)
{
    int gid = blockIdx.x * blockDim.x + threadIdx.x;
    if (gid >= Nn * OUT_C) return;
    int n = gid >> 6, c = gid & 63;
    float s = 0.f;
    #pragma unroll
    for (int h = 0; h < HEADS; ++h)
        s += acc[n * HC + h * OUT_C + c] / (ssum[n * 4 + h] + 1e-16f);
    out[gid] = 0.25f * s + b2[c] + xfc[gid];
}

// ---------------- host launcher ----------------
extern "C" void kernel_launch(void* const* d_in, const int* in_sizes, int n_in,
                              void* d_out, int out_size, void* d_ws, size_t ws_size,
                              hipStream_t stream)
{
    const float*     x      = (const float*)d_in[0];
    const long long* ei     = (const long long*)d_in[1];
    const float*     W1     = (const float*)d_in[2];
    const float*     a_src1 = (const float*)d_in[3];
    const float*     a_dst1 = (const float*)d_in[4];
    const float*     b1     = (const float*)d_in[5];
    const float*     W2     = (const float*)d_in[6];
    const float*     a_src2 = (const float*)d_in[7];
    const float*     a_dst2 = (const float*)d_in[8];
    const float*     b2     = (const float*)d_in[9];
    const float*     Wfc    = (const float*)d_in[10];
    const float*     bfc    = (const float*)d_in[11];
    float*           out    = (float*)d_out;
    (void)n_in; (void)out_size; (void)ws_size;

    const int Nn = in_sizes[0] / IN_C;
    const int E  = in_sizes[1] / 2;
    const int EN = E + Nn;

    char* ws = (char*)d_ws;
    size_t off = 0;
    auto alloc = [&](size_t bytes) -> char* {
        char* p = ws + off;
        off = (off + bytes + 255) & ~(size_t)255;
        return p;
    };
    unsigned short* xb   = (unsigned short*)alloc((size_t)Nn * IN_C * 2);
    unsigned short* hb   = (unsigned short*)alloc((size_t)Nn * HC * 2);  // h1, reused as h2
    unsigned short* f1b  = (unsigned short*)alloc((size_t)Nn * HC * 2);
    float*          acc  = (float*)alloc((size_t)Nn * HC * 4);
    unsigned*       menc = (unsigned*)alloc((size_t)Nn * HEADS * 4);
    float*          ssum = (float*)alloc((size_t)Nn * HEADS * 4);
    float*          als  = (float*)alloc((size_t)Nn * HEADS * 4);
    float*          ald  = (float*)alloc((size_t)Nn * HEADS * 4);
    float*          xfc  = (float*)alloc((size_t)Nn * OUT_C * 4);
    unsigned short* W1p  = (unsigned short*)alloc((size_t)IN_C * HC * 2);
    unsigned short* W2p  = (unsigned short*)alloc((size_t)HC * HC * 2);
    unsigned short* Wfcp = (unsigned short*)alloc((size_t)IN_C * OUT_C * 2);

    const int T = 256;
    const int gblk = (Nn + 63) / 64;

    // precision staging
    f32_to_bf16_kernel<<<(Nn * IN_C + T - 1) / T, T, 0, stream>>>(x, xb, Nn * IN_C);
    pack_b<<<(IN_C * HC + T - 1) / T, T, 0, stream>>>(W1, W1p, IN_C, HC);
    pack_b<<<(HC * HC + T - 1) / T, T, 0, stream>>>(W2, W2p, HC, HC);
    pack_b<<<(IN_C * OUT_C + T - 1) / T, T, 0, stream>>>(Wfc, Wfcp, IN_C, OUT_C);

    // h1 = x @ W1 (bias applied post-aggregation);   xfc = x @ Wfc + bfc
    gemm_wmma_bf16<IN_C, HC><<<gblk, T, 0, stream>>>(xb, W1p, nullptr, hb, nullptr, Nn);
    gemm_wmma_bf16<IN_C, OUT_C><<<gblk, T, 0, stream>>>(xb, Wfcp, bfc, nullptr, xfc, Nn);

    // ---- layer 1 attention + aggregation ----
    zero_u32<<<(Nn * HC + T - 1) / T, T, 0, stream>>>((unsigned*)acc, Nn * HC);
    zero_u32<<<(Nn * HEADS + T - 1) / T, T, 0, stream>>>(menc, Nn * HEADS);
    zero_u32<<<(Nn * HEADS + T - 1) / T, T, 0, stream>>>((unsigned*)ssum, Nn * HEADS);
    attn_scores<<<(Nn * HEADS + T - 1) / T, T, 0, stream>>>(hb, a_src1, a_dst1, als, ald, Nn);
    edge_max<<<(EN + T - 1) / T, T, 0, stream>>>(ei, E, Nn, als, ald, menc);
    edge_acc<<<(EN + 7) / 8, T, 0, stream>>>(ei, E, Nn, als, ald, menc, hb, ssum, acc);
    finalize1<<<(Nn * HC + T - 1) / T, T, 0, stream>>>(acc, ssum, b1, f1b, Nn);

    // ---- layer 2 ----
    gemm_wmma_bf16<HC, HC><<<gblk, T, 0, stream>>>(f1b, W2p, nullptr, hb, nullptr, Nn);
    zero_u32<<<(Nn * HC + T - 1) / T, T, 0, stream>>>((unsigned*)acc, Nn * HC);
    zero_u32<<<(Nn * HEADS + T - 1) / T, T, 0, stream>>>(menc, Nn * HEADS);
    zero_u32<<<(Nn * HEADS + T - 1) / T, T, 0, stream>>>((unsigned*)ssum, Nn * HEADS);
    attn_scores<<<(Nn * HEADS + T - 1) / T, T, 0, stream>>>(hb, a_src2, a_dst2, als, ald, Nn);
    edge_max<<<(EN + T - 1) / T, T, 0, stream>>>(ei, E, Nn, als, ald, menc);
    edge_acc<<<(EN + 7) / 8, T, 0, stream>>>(ei, E, Nn, als, ald, menc, hb, ssum, acc);
    finalize2<<<(Nn * OUT_C + T - 1) / T, T, 0, stream>>>(acc, ssum, b2, xfc, out, Nn);
}
